// _ScaleEncoder_60138132078601
// MI455X (gfx1250) — compile-verified, matrix operand order checked by reading, not run on hardware
//
#include <hip/hip_runtime.h>

// ---------------- problem constants (reference: B,T,D,N,L,SCALE = 16,4096,256,16,4,2) ----
#define BB 16
#define TT 4096
#define TS 2048       // pooled length (T/SCALE)
#define DD 256
#define NN 16
#define LL 4
#define RR (BB * TS)                 // 32768 rows after pooling
#define RC ((long)RR * DD)           // 8388608 elements per [R,D] tensor
#define CC 32                        // scan chunks per sequence
#define CL (TS / CC)                 // 64 steps per chunk

// ---------------- vector types for WMMA ----------------
typedef __attribute__((ext_vector_type(16))) __bf16         bf16x16;
typedef __attribute__((ext_vector_type(8)))  float          f32x8;
typedef __attribute__((ext_vector_type(8)))  unsigned short u16x8;

union Frag { bf16x16 v; u16x8 h[2]; };

// ---------------- helpers ----------------
__device__ __forceinline__ unsigned short f2bf(float f) {
  unsigned int u = __float_as_uint(f);
  u += 0x7FFFu + ((u >> 16) & 1u);           // round-to-nearest-even
  return (unsigned short)(u >> 16);
}
__device__ __forceinline__ float bf2f(unsigned short h) {
  return __uint_as_float(((unsigned int)h) << 16);
}
__device__ __forceinline__ float softplus_f(float v) {
  float r = __logf(1.f + __expf(v));
  return v > 20.f ? v : r;
}

// ---------------- weight transpose + fp32->bf16 convert: dst[n*K+k] = src[k*N+n] ----------
__global__ void tconv_k(const float* __restrict__ src, unsigned short* __restrict__ dst,
                        int K, int Nn) {
  const int i = blockIdx.x * 256 + threadIdx.x;
  if (i >= K * Nn) return;
  const int k = i / Nn, n = i - k * Nn;
  dst[(long)n * K + k] = f2bf(src[i]);
}

// ---------------- WMMA GEMM: out[M=32768, 256] = A[M,K](bf16) x W[K,256] + bias -----------
// Wt is the transposed bf16 weight [256, K]. Block = 128 thr (4 waves), tile M32 x N256.
// MODE 0: out = softplus(acc+bias)   (delta path, fp32 out)
// MODE 1: out += acc+bias            (output proj + residual, fp32 in/out)
// MODE 2: out = acc+bias             (merge, fp32 out)
template <int MODE>
__global__ __launch_bounds__(128)
void gemm_wmma(const unsigned short* __restrict__ A,
               const unsigned short* __restrict__ Wt,
               const float* __restrict__ bias,
               float* __restrict__ out,
               int K) {
  const int  lane = threadIdx.x & 31;
  const int  wave = threadIdx.x >> 5;
  const int  lm   = lane & 15;
  const int  hf   = lane >> 4;
  const long mrow = (long)blockIdx.x * 32;
  const int  nb   = wave * 64;

  f32x8 acc[2][4] = {};

  const unsigned short* ap0 = A + (mrow + lm) * (long)K;
  const unsigned short* ap1 = ap0 + 16L * K;
  const unsigned short* bp0 = Wt + (long)(nb + lm) * K;

  for (int kb = 0; kb < K; kb += 32) {
    const int ko = kb + 8 * hf;
    Frag a0, a1;
    a0.h[0] = *(const u16x8*)(ap0 + ko);
    a0.h[1] = *(const u16x8*)(ap0 + ko + 16);
    a1.h[0] = *(const u16x8*)(ap1 + ko);
    a1.h[1] = *(const u16x8*)(ap1 + ko + 16);
#pragma unroll
    for (int j = 0; j < 4; j++) {
      Frag bfr;
      const unsigned short* bp = bp0 + (long)j * 16 * K + ko;
      bfr.h[0] = *(const u16x8*)(bp);
      bfr.h[1] = *(const u16x8*)(bp + 16);
      acc[0][j] = __builtin_amdgcn_wmma_f32_16x16x32_bf16(false, a0.v, false, bfr.v,
                                                          (short)0, acc[0][j], false, false);
      acc[1][j] = __builtin_amdgcn_wmma_f32_16x16x32_bf16(false, a1.v, false, bfr.v,
                                                          (short)0, acc[1][j], false, false);
    }
  }

#pragma unroll
  for (int j = 0; j < 4; j++) {
    const int   col = nb + j * 16 + lm;
    const float bv  = bias[col];
#pragma unroll
    for (int mi = 0; mi < 2; mi++) {
#pragma unroll
      for (int r = 0; r < 8; r++) {
        const long row = mrow + mi * 16 + r + 8 * hf;
        const long o   = row * DD + col;
        const float v  = acc[mi][j][r] + bv;
        if (MODE == 0)      out[o] = softplus_f(v);
        else if (MODE == 1) out[o] += v;
        else                out[o] = v;
      }
    }
  }
}

// ---------------- LayerNorm (one wave per row of 256), writes bf16 for the GEMM A --------
__global__ __launch_bounds__(256)
void ln_k(const float* __restrict__ xr, const float* __restrict__ g,
          const float* __restrict__ bta, unsigned short* __restrict__ xn, int goff) {
  const int  wave = threadIdx.x >> 5;
  const int  lane = threadIdx.x & 31;
  const long row  = (long)blockIdx.x * 8 + wave;
  const float* p  = xr + row * DD + lane * 8;
  float v[8];
  float s = 0.f, s2 = 0.f;
#pragma unroll
  for (int i = 0; i < 8; i++) { v[i] = p[i]; s += v[i]; s2 += v[i] * v[i]; }
#pragma unroll
  for (int m = 16; m >= 1; m >>= 1) {
    s  += __shfl_xor(s, m, 32);
    s2 += __shfl_xor(s2, m, 32);
  }
  const float mean = s * (1.f / DD);
  const float var  = s2 * (1.f / DD) - mean * mean;
  const float rs   = rsqrtf(var + 1e-5f);
  unsigned short* q = xn + row * DD + lane * 8;
#pragma unroll
  for (int i = 0; i < 8; i++) {
    const int c = lane * 8 + i;
    q[i] = f2bf((v[i] - mean) * rs * g[goff + c] + bta[goff + c]);
  }
}

// ============== chunked parallel scan (3 phases) ==============
// recurrence: s_t = exp(-dt*a)*s_{t-1} + (dt*x)*b  per (b,d,n); linear, and the chunk
// transition product collapses exactly: prod_t exp(-dt_t*a) = exp(-a * sum_t dt_t).

// Pass 1: each (b, d, chunk) runs its 64-step chunk from state 0.
// Writes chunk-end state se[b][d][c][0..15] and dts[b][d][c] = sum of dt.
__global__ __launch_bounds__(256)
void scan_p1(const float* __restrict__ delta, const unsigned short* __restrict__ xn,
             const float* __restrict__ alog, const float* __restrict__ bparam,
             float* __restrict__ se, float* __restrict__ dts, int layer) {
  const int  d    = threadIdx.x;
  const int  b    = blockIdx.x >> 5;
  const int  c    = blockIdx.x & (CC - 1);
  const long base = ((long)b * TS + c * CL) * DD + d;
  const int  po   = layer * DD * NN + d * NN;
  float ap[NN], bp[NN], s[NN];
#pragma unroll
  for (int n = 0; n < NN; n++) {
    ap[n] = softplus_f(alog[po + n]) + 1e-4f;
    bp[n] = bparam[po + n];
    s[n]  = 0.f;
  }
  float dsum = 0.f;
  for (int t = 0; t < CL; t++) {
    const long idx = base + (long)t * DD;
    __builtin_prefetch(delta + idx + 4 * DD, 0, 3);
    __builtin_prefetch(xn + idx + 4 * DD, 0, 3);
    const float dt = delta[idx];
    const float dx = dt * bf2f(xn[idx]);
    dsum += dt;
#pragma unroll
    for (int n = 0; n < NN; n++) s[n] = __expf(-dt * ap[n]) * s[n] + dx * bp[n];
  }
  const long rb = (long)b * DD + d;
  const long o  = (rb * CC + c) * NN;
#pragma unroll
  for (int n = 0; n < NN; n++) se[o + n] = s[n];
  dts[rb * CC + c] = dsum;
}

// Pass 2: per (b,d), walk the 32 chunk summaries sequentially; replace each chunk-end
// state in `se` with the chunk-ENTRY state (prefix carry), in place.
__global__ __launch_bounds__(256)
void scan_p2(float* __restrict__ se, const float* __restrict__ dts,
             const float* __restrict__ alog, int layer) {
  const int  d  = threadIdx.x;
  const int  b  = blockIdx.x;
  const int  po = layer * DD * NN + d * NN;
  const long rb = (long)b * DD + d;
  float ap[NN], carry[NN];
#pragma unroll
  for (int n = 0; n < NN; n++) {
    ap[n]    = softplus_f(alog[po + n]) + 1e-4f;
    carry[n] = 0.f;
  }
  for (int c = 0; c < CC; c++) {
    const float dsum = dts[rb * CC + c];
    const long  o    = (rb * CC + c) * NN;
#pragma unroll
    for (int n = 0; n < NN; n++) {
      const float aprod = __expf(-ap[n] * dsum);
      const float endst = se[o + n];
      se[o + n] = carry[n];                       // entry state for chunk c
      carry[n]  = aprod * carry[n] + endst;
    }
  }
}

// Pass 3: re-run each chunk seeded with its entry state, emit y (bf16) for the next GEMM.
__global__ __launch_bounds__(256)
void scan_p3(const float* __restrict__ delta, const unsigned short* __restrict__ xn,
             const float* __restrict__ alog, const float* __restrict__ bparam,
             const float* __restrict__ se, unsigned short* __restrict__ y, int layer) {
  const int  d    = threadIdx.x;
  const int  b    = blockIdx.x >> 5;
  const int  c    = blockIdx.x & (CC - 1);
  const long base = ((long)b * TS + c * CL) * DD + d;
  const int  po   = layer * DD * NN + d * NN;
  const long o    = (((long)b * DD + d) * CC + c) * NN;
  float ap[NN], bp[NN], s[NN];
#pragma unroll
  for (int n = 0; n < NN; n++) {
    ap[n] = softplus_f(alog[po + n]) + 1e-4f;
    bp[n] = bparam[po + n];
    s[n]  = se[o + n];
  }
  for (int t = 0; t < CL; t++) {
    const long idx = base + (long)t * DD;
    __builtin_prefetch(delta + idx + 4 * DD, 0, 3);
    __builtin_prefetch(xn + idx + 4 * DD, 0, 3);
    const float dt = delta[idx];
    const float dx = dt * bf2f(xn[idx]);
    float ysum = 0.f;
#pragma unroll
    for (int n = 0; n < NN; n++) {
      s[n] = __expf(-dt * ap[n]) * s[n] + dx * bp[n];
      ysum += s[n];
    }
    y[idx] = f2bf(ysum);
  }
}

// ---------------- elementwise kernels ----------------
__global__ void pool_k(const float* __restrict__ x, float* __restrict__ xs) {
  const long i = (long)blockIdx.x * 256 + threadIdx.x;
  const int  d = (int)(i & 255);
  const int  t = (int)((i >> 8) & (TS - 1));
  const int  b = (int)(i >> 19);
  const float* p = x + (((long)b * TT + 2 * t) * DD + d);
  xs[i] = 0.5f * (p[0] + p[DD]);
}

__global__ void copy_k(const float* __restrict__ s, float* __restrict__ dst) {
  const long i = (long)blockIdx.x * 256 + threadIdx.x;
  dst[i] = s[i];
}

__global__ void flip_k(const float* __restrict__ xs, float* __restrict__ res) {
  const long i = (long)blockIdx.x * 256 + threadIdx.x;
  const int  d = (int)(i & 255);
  const int  t = (int)((i >> 8) & (TS - 1));
  const int  b = (int)(i >> 19);
  res[i] = xs[((long)b * TS + (TS - 1 - t)) * DD + d];
}

// pack one 256-wide half of the 512-wide merge input; dir==1 un-flips the bwd result
__global__ void half_k(const float* __restrict__ res, unsigned short* __restrict__ mi, int dir) {
  const long i = (long)blockIdx.x * 256 + threadIdx.x;
  const int  d = (int)(i & 255);
  const int  t = (int)((i >> 8) & (TS - 1));
  const int  b = (int)(i >> 19);
  const long src = dir ? ((long)b * TS + (TS - 1 - t)) * DD + d : i;
  mi[((long)b * TS + t) * (2 * DD) + dir * DD + d] = f2bf(res[src]);
}

__global__ void ups_k(const float* __restrict__ mo, float* __restrict__ out) {
  const long i = (long)blockIdx.x * 256 + threadIdx.x;   // over B*T*D = 2^24
  const int  d = (int)(i & 255);
  const int  t = (int)((i >> 8) & (TT - 1));
  const int  b = (int)(i >> 20);
  float pos = (t + 0.5f) * ((float)TS / (float)TT) - 0.5f;
  pos = fminf(fmaxf(pos, 0.f), (float)(TS - 1));
  const int   lo = (int)floorf(pos);
  const int   hi = min(lo + 1, TS - 1);
  const float w  = pos - (float)lo;
  const long  base = (long)b * TS * DD + d;
  out[i] = mo[base + (long)lo * DD] * (1.f - w) + mo[base + (long)hi * DD] * w;
}

// ---------------- host orchestration ----------------
extern "C" void kernel_launch(void* const* d_in, const int* in_sizes, int n_in,
                              void* d_out, int out_size, void* d_ws, size_t ws_size,
                              hipStream_t stream) {
  (void)in_sizes; (void)n_in; (void)out_size; (void)ws_size;

  const float* x       = (const float*)d_in[0];
  const float* nw[2]   = {(const float*)d_in[1],  (const float*)d_in[9]};
  const float* nbp[2]  = {(const float*)d_in[2],  (const float*)d_in[10]};
  const float* dw[2]   = {(const float*)d_in[3],  (const float*)d_in[11]};
  const float* db[2]   = {(const float*)d_in[4],  (const float*)d_in[12]};
  const float* alog[2] = {(const float*)d_in[5],  (const float*)d_in[13]};
  const float* bpar[2] = {(const float*)d_in[6],  (const float*)d_in[14]};
  const float* ow[2]   = {(const float*)d_in[7],  (const float*)d_in[15]};
  const float* ob[2]   = {(const float*)d_in[8],  (const float*)d_in[16]};
  const float* merge_w = (const float*)d_in[17];
  const float* merge_b = (const float*)d_in[18];

  char* ws = (char*)d_ws;
  float*          RES   = (float*)(ws + 0);                       // 32 MB residual
  unsigned short* XN    = (unsigned short*)(ws + 33554432);       // 16 MB LN out (bf16)
  float*          DELTA = (float*)(ws + 50331648);                // 32 MB delta
  unsigned short* Y     = (unsigned short*)(ws + 83886080);       // 16 MB scan out (bf16)
  unsigned short* MI    = (unsigned short*)(ws + 100663296);      // 32 MB merge input (bf16)
  float*          MO    = (float*)(ws + 134217728);               // 32 MB merged out / pooled XS
  float*          XS    = MO;                                     // alias (disjoint lifetimes)
  unsigned short* WTB   = (unsigned short*)(ws + 167772160);      // ~2.25 MB bf16 weights^T
  float*          SE    = (float*)(ws + 170131456);               // 8 MB chunk states [B,D,CC,NN]
  float*          DTS   = (float*)(ws + 178520064);               // 0.5 MB chunk dt-sums [B,D,CC]

  const long MAT = (long)DD * DD;   // 65536 elements per square weight

  // 1) weights: transpose + bf16 convert (stay hot in 192 MB L2 afterwards)
  for (int dir = 0; dir < 2; dir++) {
    for (int l = 0; l < LL; l++) {
      tconv_k<<<256, 256, 0, stream>>>(dw[dir] + l * MAT, WTB + (dir * 4 + l) * MAT, DD, DD);
      tconv_k<<<256, 256, 0, stream>>>(ow[dir] + l * MAT, WTB + (8 + dir * 4 + l) * MAT, DD, DD);
    }
  }
  tconv_k<<<512, 256, 0, stream>>>(merge_w, WTB + 16 * MAT, 2 * DD, DD);

  // 2) average-pool T 4096 -> 2048
  pool_k<<<RC / 256, 256, 0, stream>>>(x, XS);

  // 3) both directions through the 4-layer stack
  for (int dir = 0; dir < 2; dir++) {
    if (dir == 0) copy_k<<<RC / 256, 256, 0, stream>>>(XS, RES);
    else          flip_k<<<RC / 256, 256, 0, stream>>>(XS, RES);

    for (int l = 0; l < LL; l++) {
      ln_k<<<RR / 8, 256, 0, stream>>>(RES, nw[dir], nbp[dir], XN, l * DD);
      gemm_wmma<0><<<RR / 32, 128, 0, stream>>>(XN, WTB + (dir * 4 + l) * MAT,
                                                db[dir] + l * DD, DELTA, DD);
      // chunked parallel scan: 4096 waves instead of 128
      scan_p1<<<BB * CC, DD, 0, stream>>>(DELTA, XN, alog[dir], bpar[dir], SE, DTS, l);
      scan_p2<<<BB, DD, 0, stream>>>(SE, DTS, alog[dir], l);
      scan_p3<<<BB * CC, DD, 0, stream>>>(DELTA, XN, alog[dir], bpar[dir], SE, Y, l);
      gemm_wmma<1><<<RR / 32, 128, 0, stream>>>(Y, WTB + (8 + dir * 4 + l) * MAT,
                                                ob[dir] + l * DD, RES, DD);
    }
    half_k<<<RC / 256, 256, 0, stream>>>(RES, MI, dir);
  }

  // 4) merge GEMM: [R,512] x [512,256]
  gemm_wmma<2><<<RR / 32, 128, 0, stream>>>(MI, WTB + 16 * MAT, merge_b, MO, 2 * DD);

  // 5) linear upsample 2048 -> 4096
  ups_k<<<(long)BB * TT * DD / 256, 256, 0, stream>>>(MO, (float*)d_out);
}
